// GraphTransformerNet_46248207843787
// MI455X (gfx1250) — compile-verified, matrix operand order checked by reading, not run on hardware
//
#include <hip/hip_runtime.h>
#include <math.h>

// ---------------- problem constants (match reference) ----------------
#define Bg   64
#define NNg  256
#define Ng   (Bg * NNg)     // 16384 nodes
#define DEGg 8
#define Eg   (Ng * DEGg)    // 131072 edges
#define Dg   512
#define Hg   8
#define DHg  64
#define Lg   6
#define DFFg 1024

// ---------------- WMMA types (CDNA5 / gfx1250, wave32) ----------------
typedef __attribute__((ext_vector_type(16))) __bf16 v16bf;
typedef __attribute__((ext_vector_type(8)))  float  v8f;
typedef __attribute__((ext_vector_type(4)))  int    v4i_a;   // async-to-LDS payload type

union ABFrag { v16bf v; uint4 q[2]; };   // 32 bytes: one 16x32 bf16 fragment per lane

// gfx1250 async memory->LDS path (ASYNCcnt), guarded so both host pass and
// older device toolchains fall back to plain LDS stores.
#if defined(__has_builtin)
#if __has_builtin(__builtin_amdgcn_global_load_async_to_lds_b128) && \
    __has_builtin(__builtin_amdgcn_s_wait_asynccnt)
#define USE_ASYNC_LDS 1
#endif
#endif
#ifndef USE_ASYNC_LDS
#define USE_ASYNC_LDS 0
#endif

// float -> bf16 (round to nearest even), stored as raw ushort
static __device__ __forceinline__ unsigned short f2bf(float f) {
    union { float f; unsigned int u; } x; x.f = f;
    unsigned int u = x.u;
    unsigned int r = (u + 0x7FFFu + ((u >> 16) & 1u)) >> 16;
    return (unsigned short)r;
}

// stage 32 contiguous bytes global -> LDS (per thread)
static __device__ __forceinline__ void stage32(const unsigned short* g, unsigned short* l) {
#if USE_ASYNC_LDS
    __builtin_amdgcn_global_load_async_to_lds_b128(
        (__attribute__((address_space(1))) v4i_a*)g,
        (__attribute__((address_space(3))) v4i_a*)l, 0, 0);
    __builtin_amdgcn_global_load_async_to_lds_b128(
        (__attribute__((address_space(1))) v4i_a*)g,
        (__attribute__((address_space(3))) v4i_a*)l, 16, 0);
#else
    uint4 d0 = *(const uint4*)(g);
    uint4 d1 = *(const uint4*)(g + 8);
    *(uint4*)(l)     = d0;
    *(uint4*)(l + 8) = d1;
#endif
}

static __device__ __forceinline__ void stage_wait() {
#if USE_ASYNC_LDS
    __builtin_amdgcn_s_wait_asynccnt(0);
#endif
}

// ---------------- small-K input embedding: out = X @ W + b ----------------
__global__ void embed_kernel(const float* __restrict__ X, const float* __restrict__ W,
                             const float* __restrict__ b, float* __restrict__ out,
                             int M, int KK, int Dd) {
    size_t i = (size_t)blockIdx.x * blockDim.x + threadIdx.x;
    if (i >= (size_t)M * Dd) return;
    int row = (int)(i / Dd), col = (int)(i % Dd);
    float a = b[col];
    const float* xr = X + (size_t)row * KK;
    for (int k = 0; k < KK; ++k) a += xr[k] * W[(size_t)k * Dd + col];
    out[i] = a;
}

// ---------------- f32 -> bf16(ushort) conversion pass ----------------
__global__ void cvt_bf16_kernel(const float* __restrict__ X, unsigned short* __restrict__ Y, size_t n) {
    size_t i = (size_t)blockIdx.x * blockDim.x + threadIdx.x;
    if (i < n) Y[i] = f2bf(X[i]);
}

__global__ void fill0_kernel(float* __restrict__ p, size_t n) {
    size_t i = (size_t)blockIdx.x * blockDim.x + threadIdx.x;
    if (i < n) p[i] = 0.0f;
}

// ---------------- weight packing ----------------
// Pack f32 W[K x Nc] into bf16 fragment-major layout:
//   P[((nt*ktiles + kt)*32 + lane)*16 + elem]
// so a lane's whole 16x32 B-fragment (B = 32x16 tile, lane -> N, elems -> K per
// the CDNA5 16-bit operand layout) is one contiguous 32-byte read.
__global__ void pack_w_kernel(const float* __restrict__ W, unsigned short* __restrict__ P,
                              int K, int Nc) {
    size_t i = (size_t)blockIdx.x * blockDim.x + threadIdx.x;
    if (i >= (size_t)K * Nc) return;
    int elem = (int)(i & 15);
    int lane = (int)((i >> 4) & 31);
    size_t tile = i >> 9;
    int ktiles = K >> 5;
    int kt = (int)(tile % ktiles);
    int nt = (int)(tile / ktiles);
    int hv = lane >> 4;                 // half-wave selects K group
    int n  = nt * 16 + (lane & 15);
    int v = elem >> 1, pb = elem & 1;
    int k;
    if (v < 4) k = kt * 32 + hv * 8 + 2 * v + pb;
    else       k = kt * 32 + 16 + hv * 8 + 2 * (v - 4) + pb;
    P[i] = f2bf(W[(size_t)k * Nc + n]);
}

// ---------------- core tiled BF16 WMMA GEMM ----------------
// C[M x Nc] = act(A_bf16[M x K] @ Wpacked[K x Nc] + bias [+ resid]) -> f32 and/or bf16
// block tile 128x128, 8 waves (4 in M x 2 in N), wave tile 32x64 (2x4 fragments),
// K step 32, double-buffered LDS A-tile staged via async-to-LDS (one barrier/K-step).
#define LDS_STRIDE 40   // 32 + 8 pad elems -> conflict-free ds_load_b128 across 16 rows

template <bool RELU>
__global__ __launch_bounds__(256)
void wmma_gemm_kernel(const unsigned short* __restrict__ A,
                      const unsigned short* __restrict__ Wp,
                      const float* __restrict__ bias,
                      const float* __restrict__ resid,
                      float* __restrict__ outF,
                      unsigned short* __restrict__ outB,
                      int M, int K, int Nc) {
    __shared__ unsigned short sA[2][128 * LDS_STRIDE];
    const int tid   = threadIdx.x;
    const int wave  = tid >> 5;
    const int lane  = tid & 31;
    const int waveM = wave & 3;          // * 32 rows
    const int waveN = wave >> 2;         // * 64 cols
    const int blockM = blockIdx.x * 128;
    const int blockN = blockIdx.y * 128;
    const int hv  = lane >> 4;
    const int l16 = lane & 15;
    const int ktiles = K >> 5;

    // staging: each thread copies 32B of the 128x32 bf16 A tile per K-step
    const int sr = tid >> 1, sseg = tid & 1;
    const unsigned short* gstage = A + (size_t)(blockM + sr) * K + sseg * 16;
    unsigned short* lst[2] = { &sA[0][sr * LDS_STRIDE + sseg * 16],
                               &sA[1][sr * LDS_STRIDE + sseg * 16] };

    // A fragment base pointers (per buffer / mi); second 16B half lives at +16 elems
    const unsigned short* afb[2][2];
    for (int b2 = 0; b2 < 2; ++b2)
        for (int mi = 0; mi < 2; ++mi)
            afb[b2][mi] = &sA[b2][(waveM * 32 + mi * 16 + l16) * LDS_STRIDE + hv * 8];

    // B fragment pointers: advance by one packed k-tile (512 elems = 1KB) per step
    const unsigned short* bptr[4];
    for (int ni = 0; ni < 4; ++ni) {
        int nt = (blockN >> 4) + waveN * 4 + ni;
        bptr[ni] = Wp + ((size_t)nt * ktiles * 32 + lane) * 16;
    }

    v8f acc[2][4] = {};

    // prologue: stage tile 0
    stage32(gstage, lst[0]);
    gstage += 32;
    stage_wait();
    __syncthreads();

    for (int kt = 0; kt < ktiles; ++kt) {
        const int cur = kt & 1;
        if (kt + 1 < ktiles) {            // stage next tile into the other buffer
            stage32(gstage, lst[cur ^ 1]);
            gstage += 32;
            __builtin_prefetch(bptr[0] + 512, 0, 1);   // next packed-B k-tile
        }

        ABFrag af[2];
        for (int mi = 0; mi < 2; ++mi) {
            af[mi].q[0] = *(const uint4*)(afb[cur][mi]);
            af[mi].q[1] = *(const uint4*)(afb[cur][mi] + 16);
        }
        ABFrag bfr[4];
        for (int ni = 0; ni < 4; ++ni) {
            bfr[ni].q[0] = *(const uint4*)(bptr[ni]);
            bfr[ni].q[1] = *(const uint4*)(bptr[ni] + 8);
            bptr[ni] += 512;
        }
        for (int mi = 0; mi < 2; ++mi)
            for (int ni = 0; ni < 4; ++ni)
                acc[mi][ni] = __builtin_amdgcn_wmma_f32_16x16x32_bf16(
                    false, af[mi].v, false, bfr[ni].v, (short)0, acc[mi][ni], false, false);

        if (kt + 1 < ktiles) {
            stage_wait();                 // our async writes to LDS landed
            __syncthreads();              // visible to all waves
        }
    }

    // epilogue: D layout — VGPR v -> M = v + 8*half, lane&15 -> N
    for (int mi = 0; mi < 2; ++mi) {
        for (int ni = 0; ni < 4; ++ni) {
            int col = blockN + waveN * 64 + ni * 16 + l16;
            float bv = bias ? bias[col] : 0.0f;
            for (int v = 0; v < 8; ++v) {
                int row = blockM + waveM * 32 + mi * 16 + v + hv * 8;
                float x = acc[mi][ni][v] + bv;
                if (resid) x += resid[(size_t)row * Nc + col];
                if (RELU)  x = fmaxf(x, 0.0f);
                if (outF) outF[(size_t)row * Nc + col] = x;
                if (outB) outB[(size_t)row * Nc + col] = f2bf(x);
            }
        }
    }
}

// ---------------- attention edge kernel: one wave per (edge, head) ----------------
__global__ __launch_bounds__(256)
void attn_edge_kernel(const float* __restrict__ Q, const float* __restrict__ Km,
                      const float* __restrict__ V, const float* __restrict__ pe,
                      const int* __restrict__ src, const int* __restrict__ dst,
                      float* __restrict__ eout, float* __restrict__ wV, float* __restrict__ z) {
    int gw   = (int)((blockIdx.x * (size_t)blockDim.x + threadIdx.x) >> 5);
    int lane = threadIdx.x & 31;
    int eidx = gw >> 3;       // H = 8
    int hh   = gw & 7;
    if (eidx >= Eg) return;
    int s = src[eidx], d = dst[eidx];
    size_t baseS = (size_t)s * Dg + hh * DHg;
    size_t baseD = (size_t)d * Dg + hh * DHg;
    size_t baseE = (size_t)eidx * Dg + hh * DHg;
    const float scale = 0.125f;            // 1/sqrt(64)
    float partial = 0.0f;
    for (int t = 0; t < 2; ++t) {
        int dh = lane + t * 32;
        float v = Km[baseS + dh] * Q[baseD + dh] * scale * pe[baseE + dh];
        eout[baseE + dh] = v;
        partial += v;
    }
    for (int off = 16; off; off >>= 1) partial += __shfl_xor(partial, off, 32);
    float sg = __expf(fminf(fmaxf(partial, -5.0f), 5.0f));
    for (int t = 0; t < 2; ++t) {
        int dh = lane + t * 32;
        atomicAdd(&wV[baseD + dh], sg * V[baseS + dh]);
    }
    if (lane == 0) atomicAdd(&z[(size_t)d * Hg + hh], sg);
}

// h_attn = wV / (z + 1e-6), in place on wV
__global__ void attn_norm_kernel(float* __restrict__ wV, const float* __restrict__ z) {
    size_t i = (size_t)blockIdx.x * blockDim.x + threadIdx.x;
    if (i >= (size_t)Ng * Dg) return;
    int row = (int)(i >> 9), col = (int)(i & 511);
    wV[i] = wV[i] / (z[(size_t)row * Hg + (col >> 6)] + 1e-6f);
}

// ---------------- BatchNorm: column sums/sumsq + apply ----------------
__global__ void colstats_kernel(const float* __restrict__ X, float* __restrict__ sums,
                                int M, int Ncols, int rowsPerBlock) {
    int col = blockIdx.x * blockDim.x + threadIdx.x;
    if (col >= Ncols) return;
    int r0 = blockIdx.y * rowsPerBlock;
    int r1 = r0 + rowsPerBlock; if (r1 > M) r1 = M;
    float s = 0.0f, s2 = 0.0f;
    for (int r = r0; r < r1; ++r) {
        float v = X[(size_t)r * Ncols + col];
        s += v; s2 += v * v;
    }
    atomicAdd(&sums[col], s);
    atomicAdd(&sums[Ncols + col], s2);
}

__global__ void bn_apply_kernel(const float* __restrict__ X, const float* __restrict__ sums,
                                const float* __restrict__ g, const float* __restrict__ b,
                                float* __restrict__ Y, int M, int Ncols) {
    size_t i = (size_t)blockIdx.x * blockDim.x + threadIdx.x;
    if (i >= (size_t)M * Ncols) return;
    int col = (int)(i % Ncols);
    float mu  = sums[col] / (float)M;
    float var = sums[Ncols + col] / (float)M - mu * mu;
    Y[i] = g[col] * (X[i] - mu) * rsqrtf(var + 1e-5f) + b[col];
}

// ---------------- final readout ----------------
// pairs[i] = concat(h[veh_global(b)], h[i]) as bf16
__global__ void pairs_kernel(const float* __restrict__ h, const int* __restrict__ veh,
                             unsigned short* __restrict__ P) {
    size_t i = (size_t)blockIdx.x * blockDim.x + threadIdx.x;
    if (i >= (size_t)Ng * (2 * Dg)) return;
    int row = (int)(i >> 10), col = (int)(i & 1023);
    int b = row >> 8;                          // NN = 256
    float v;
    if (col < Dg) {
        int ks = b * NNg + veh[b];
        v = h[(size_t)ks * Dg + col];
    } else {
        v = h[(size_t)row * Dg + (col - Dg)];
    }
    P[i] = f2bf(v);
}

__global__ void final_dot_kernel(const float* __restrict__ X, const float* __restrict__ w,
                                 const float* __restrict__ b, float* __restrict__ out) {
    int i = blockIdx.x * blockDim.x + threadIdx.x;
    if (i >= Ng) return;
    float a = b[0];
    const float* xr = X + (size_t)i * DFFg;
    for (int k = 0; k < DFFg; ++k) a += xr[k] * w[k];
    out[i] = a;
}

// ---------------- host-side launch helpers ----------------
static void launch_gemm(const unsigned short* A, const unsigned short* Wp,
                        const float* bias, const float* resid,
                        float* outF, unsigned short* outB,
                        int M, int K, int Nc, bool relu, hipStream_t st) {
    dim3 grid(M / 128, Nc / 128), blk(256);
    if (relu) wmma_gemm_kernel<true ><<<grid, blk, 0, st>>>(A, Wp, bias, resid, outF, outB, M, K, Nc);
    else      wmma_gemm_kernel<false><<<grid, blk, 0, st>>>(A, Wp, bias, resid, outF, outB, M, K, Nc);
}

static void launch_pack(const float* W, unsigned short* P, int K, int Nc, hipStream_t st) {
    size_t n = (size_t)K * Nc;
    pack_w_kernel<<<(unsigned)((n + 255) / 256), 256, 0, st>>>(W, P, K, Nc);
}

static void launch_cvt(const float* X, unsigned short* Y, size_t n, hipStream_t st) {
    cvt_bf16_kernel<<<(unsigned)((n + 255) / 256), 256, 0, st>>>(X, Y, n);
}

static void launch_fill0(float* p, size_t n, hipStream_t st) {
    fill0_kernel<<<(unsigned)((n + 255) / 256), 256, 0, st>>>(p, n);
}

static void launch_bn(float* X, float* Y, const float* g, const float* b,
                      float* stats, int M, int Ncols, hipStream_t st) {
    launch_fill0(stats, 2 * (size_t)Ncols, st);
    dim3 sg((Ncols + 255) / 256, M / 256);
    colstats_kernel<<<sg, 256, 0, st>>>(X, stats, M, Ncols, 256);
    size_t n = (size_t)M * Ncols;
    bn_apply_kernel<<<(unsigned)((n + 255) / 256), 256, 0, st>>>(X, stats, g, b, Y, M, Ncols);
}

// ---------------- main entry ----------------
extern "C" void kernel_launch(void* const* d_in, const int* in_sizes, int n_in,
                              void* d_out, int out_size, void* d_ws, size_t ws_size,
                              hipStream_t stream) {
    (void)in_sizes; (void)n_in; (void)out_size; (void)ws_size;
    // Inputs in setup_inputs() dict order (params flattened in insertion order).
    const float* h_in  = (const float*)d_in[0];
    const float* e_in  = (const float*)d_in[1];
    const int*   src   = (const int*)d_in[2];
    const int*   dst   = (const int*)d_in[3];
    const int*   veh   = (const int*)d_in[4];
    // d_in[5] = num_nodes scalar (== NNg)
    const float* Wh  = (const float*)d_in[6];
    const float* bh  = (const float*)d_in[7];
    const float* We  = (const float*)d_in[8];
    const float* be  = (const float*)d_in[9];
    const float* Wq   = (const float*)d_in[10];  // [L,512,512]
    const float* Wk   = (const float*)d_in[11];
    const float* Wv   = (const float*)d_in[12];
    const float* Wpe  = (const float*)d_in[13];
    const float* Woh  = (const float*)d_in[14];
    const float* boh  = (const float*)d_in[15];  // [L,512]
    const float* Woe  = (const float*)d_in[16];
    const float* boe  = (const float*)d_in[17];
    const float* Wf1h = (const float*)d_in[18];  // [L,512,1024]
    const float* bf1h = (const float*)d_in[19];
    const float* Wf2h = (const float*)d_in[20];  // [L,1024,512]
    const float* bf2h = (const float*)d_in[21];
    const float* Wf1e = (const float*)d_in[22];
    const float* bf1e = (const float*)d_in[23];
    const float* Wf2e = (const float*)d_in[24];
    const float* bf2e = (const float*)d_in[25];
    const float* g1h  = (const float*)d_in[26];
    const float* g1e  = (const float*)d_in[27];
    const float* g2h  = (const float*)d_in[28];
    const float* g2e  = (const float*)d_in[29];
    const float* b1h  = (const float*)d_in[30];
    const float* b1e  = (const float*)d_in[31];
    const float* b2h  = (const float*)d_in[32];
    const float* b2e  = (const float*)d_in[33];
    const float* Wm1  = (const float*)d_in[34];
    const float* bm1  = (const float*)d_in[35];
    const float* Wm2  = (const float*)d_in[36];
    const float* bm2  = (const float*)d_in[37];
    const float* Wm3  = (const float*)d_in[38];
    const float* bm3  = (const float*)d_in[39];
    float* out = (float*)d_out;

    // ---- workspace layout ----
    char* ws = (char*)d_ws;
    size_t off = 0;
    auto alloc = [&](size_t bytes) -> char* {
        char* p = ws + off;
        off += (bytes + 255) & ~(size_t)255;
        return p;
    };
    const size_t ND = (size_t)Ng * Dg, ED = (size_t)Eg * Dg;
    float* h0  = (float*)alloc(ND * 4);
    float* Qb  = (float*)alloc(ND * 4);
    float* Kb  = (float*)alloc(ND * 4);
    float* Vb  = (float*)alloc(ND * 4);
    float* wV  = (float*)alloc(ND * 4);
    float* h2  = (float*)alloc(ND * 4);
    float* z   = (float*)alloc((size_t)Ng * Hg * 4);
    float* e0   = (float*)alloc(ED * 4);
    float* peb  = (float*)alloc(ED * 4);
    float* eoutb= (float*)alloc(ED * 4);
    float* e2   = (float*)alloc(ED * 4);
    float* stats = (float*)alloc(2 * 1024 * 4);
    unsigned short* hbf    = (unsigned short*)alloc(ND * 2);
    unsigned short* ebf    = (unsigned short*)alloc(ED * 2);
    unsigned short* hmidbf = (unsigned short*)alloc((size_t)Ng * 2 * Dg * 2); // also pairs buf
    unsigned short* emidbf = (unsigned short*)alloc((size_t)Eg * 2 * Dg * 2); // also mlp1 buf
    // packed weights
    const size_t SQ = (size_t)Dg * Dg;          // 262144
    const size_t SF = (size_t)Dg * 2 * Dg;      // 524288
    const size_t PL = 6 * SQ + 4 * SF;          // per-layer packed elems
    unsigned short* packs = (unsigned short*)alloc((PL * Lg + 2 * (size_t)DFFg * DFFg) * 2);
    unsigned short* Wm1P = packs + PL * Lg;
    unsigned short* Wm2P = Wm1P + (size_t)DFFg * DFFg;

    // ---- pack all weights (each call; deterministic) ----
    for (int l = 0; l < Lg; ++l) {
        unsigned short* pb = packs + (size_t)l * PL;
        launch_pack(Wq  + (size_t)l * SQ, pb + 0 * SQ, Dg, Dg, stream);
        launch_pack(Wk  + (size_t)l * SQ, pb + 1 * SQ, Dg, Dg, stream);
        launch_pack(Wv  + (size_t)l * SQ, pb + 2 * SQ, Dg, Dg, stream);
        launch_pack(Wpe + (size_t)l * SQ, pb + 3 * SQ, Dg, Dg, stream);
        launch_pack(Woh + (size_t)l * SQ, pb + 4 * SQ, Dg, Dg, stream);
        launch_pack(Woe + (size_t)l * SQ, pb + 5 * SQ, Dg, Dg, stream);
        launch_pack(Wf1h + (size_t)l * SF, pb + 6 * SQ + 0 * SF, Dg, 2 * Dg, stream);
        launch_pack(Wf2h + (size_t)l * SF, pb + 6 * SQ + 1 * SF, 2 * Dg, Dg, stream);
        launch_pack(Wf1e + (size_t)l * SF, pb + 6 * SQ + 2 * SF, Dg, 2 * Dg, stream);
        launch_pack(Wf2e + (size_t)l * SF, pb + 6 * SQ + 3 * SF, 2 * Dg, Dg, stream);
    }
    launch_pack(Wm1, Wm1P, DFFg, DFFg, stream);
    launch_pack(Wm2, Wm2P, DFFg, DFFg, stream);

    // ---- input embeddings ----
    embed_kernel<<<(unsigned)((ND + 255) / 256), 256, 0, stream>>>(h_in, Wh, bh, h0, Ng, 16, Dg);
    embed_kernel<<<(unsigned)((ED + 255) / 256), 256, 0, stream>>>(e_in, We, be, e0, Eg, 8, Dg);

    // ---- transformer layers ----
    for (int l = 0; l < Lg; ++l) {
        unsigned short* pb = packs + (size_t)l * PL;
        unsigned short *pWq = pb, *pWk = pb + SQ, *pWv = pb + 2 * SQ, *pWpe = pb + 3 * SQ;
        unsigned short *pWoh = pb + 4 * SQ, *pWoe = pb + 5 * SQ;
        unsigned short *pWf1h = pb + 6 * SQ, *pWf2h = pb + 6 * SQ + SF;
        unsigned short *pWf1e = pb + 6 * SQ + 2 * SF, *pWf2e = pb + 6 * SQ + 3 * SF;
        const float *boh_l = boh + (size_t)l * Dg,  *boe_l = boe + (size_t)l * Dg;
        const float *bf1h_l = bf1h + (size_t)l * 2 * Dg, *bf2h_l = bf2h + (size_t)l * Dg;
        const float *bf1e_l = bf1e + (size_t)l * 2 * Dg, *bf2e_l = bf2e + (size_t)l * Dg;
        const float *g1h_l = g1h + (size_t)l * Dg, *b1h_l = b1h + (size_t)l * Dg;
        const float *g1e_l = g1e + (size_t)l * Dg, *b1e_l = b1e + (size_t)l * Dg;
        const float *g2h_l = g2h + (size_t)l * Dg, *b2h_l = b2h + (size_t)l * Dg;
        const float *g2e_l = g2e + (size_t)l * Dg, *b2e_l = b2e + (size_t)l * Dg;

        // Q/K/V projections
        launch_cvt(h0, hbf, ND, stream);
        launch_gemm(hbf, pWq, nullptr, nullptr, Qb, nullptr, Ng, Dg, Dg, false, stream);
        launch_gemm(hbf, pWk, nullptr, nullptr, Kb, nullptr, Ng, Dg, Dg, false, stream);
        launch_gemm(hbf, pWv, nullptr, nullptr, Vb, nullptr, Ng, Dg, Dg, false, stream);
        // edge pe projection
        launch_cvt(e0, ebf, ED, stream);
        launch_gemm(ebf, pWpe, nullptr, nullptr, peb, nullptr, Eg, Dg, Dg, false, stream);
        // attention: segment softmax-weighted aggregation
        launch_fill0(wV, ND, stream);
        launch_fill0(z, (size_t)Ng * Hg, stream);
        {
            size_t thr = (size_t)Eg * Hg * 32;
            attn_edge_kernel<<<(unsigned)(thr / 256), 256, 0, stream>>>(Qb, Kb, Vb, peb, src, dst, eoutb, wV, z);
            attn_norm_kernel<<<(unsigned)((ND + 255) / 256), 256, 0, stream>>>(wV, z);
        }
        // output projections + residual
        launch_cvt(wV, hbf, ND, stream);
        launch_gemm(hbf, pWoh, boh_l, h0, h2, nullptr, Ng, Dg, Dg, false, stream);
        launch_cvt(eoutb, ebf, ED, stream);
        launch_gemm(ebf, pWoe, boe_l, e0, e2, nullptr, Eg, Dg, Dg, false, stream);
        // BN1 (in place)
        launch_bn(h2, h2, g1h_l, b1h_l, stats, Ng, Dg, stream);
        launch_bn(e2, e2, g1e_l, b1e_l, stats, Eg, Dg, stream);
        // FFN h: h3 = h2 + relu(h2@W1+b1)@W2+b2 ; BN2 -> next h0
        launch_cvt(h2, hbf, ND, stream);
        launch_gemm(hbf, pWf1h, bf1h_l, nullptr, nullptr, hmidbf, Ng, Dg, 2 * Dg, true, stream);
        launch_gemm(hmidbf, pWf2h, bf2h_l, h2, Qb, nullptr, Ng, 2 * Dg, Dg, false, stream);
        launch_bn(Qb, h0, g2h_l, b2h_l, stats, Ng, Dg, stream);
        // FFN e -> next e0
        launch_cvt(e2, ebf, ED, stream);
        launch_gemm(ebf, pWf1e, bf1e_l, nullptr, nullptr, emidbf, Eg, Dg, 2 * Dg, true, stream);
        launch_gemm(emidbf, pWf2e, bf2e_l, e2, peb, nullptr, Eg, 2 * Dg, Dg, false, stream);
        launch_bn(peb, e0, g2e_l, b2e_l, stats, Eg, Dg, stream);
    }

    // ---- readout MLP ----
    unsigned short* pairsbf = hmidbf;                 // N x 1024 bf16
    unsigned short* mlp1bf  = emidbf;                 // N x 1024 bf16 (reuse)
    float*          mlp2f   = e2;                     // N x 1024 f32 (reuse)
    {
        size_t n = (size_t)Ng * 2 * Dg;
        pairs_kernel<<<(unsigned)((n + 255) / 256), 256, 0, stream>>>(h0, veh, pairsbf);
    }
    launch_gemm(pairsbf, Wm1P, bm1, nullptr, nullptr, mlp1bf, Ng, 2 * Dg, DFFg, true, stream);
    launch_gemm(mlp1bf,  Wm2P, bm2, nullptr, mlp2f, nullptr, Ng, DFFg, DFFg, true, stream);
    final_dot_kernel<<<(Ng + 255) / 256, 256, 0, stream>>>(mlp2f, Wm3, bm3, out);
}
